// ScaledDotProductAttention_46402826666330
// MI455X (gfx1250) — matrix operand order repaired and checked by reading, not compiled, not run on hardware
//
#include <hip/hip_runtime.h>
#include <math.h>

// Problem constants from the reference: Q,K,V: [B,H,S,D] fp32.
#define B_ 4
#define H_ 16
#define S_ 2048
#define D_ 128

#define KBLK 64            // keys per iteration (2 x WMMA-K for the PV GEMM)
#define WAVES 8            // 256 threads / wave32
#define QROWS 16           // query rows per wave (WMMA M)
#define QBLK (WAVES * QROWS)

typedef __attribute__((ext_vector_type(16))) __bf16 v16bf;
typedef __attribute__((ext_vector_type(8)))  float  v8f;
typedef __attribute__((ext_vector_type(4)))  float  v4f;

__device__ __forceinline__ float fast_exp2(float x) {
#if __has_builtin(__builtin_amdgcn_exp2f)
    return __builtin_amdgcn_exp2f(x);   // v_exp_f32 is base-2 natively
#else
    return exp2f(x);
#endif
}

// Max-reduce across each 16-lane half (row group of the WMMA C layout).
__device__ __forceinline__ float halfgroup_max(float x) {
#if __has_builtin(__builtin_amdgcn_permlane16)
    unsigned u, t;
    u = __float_as_uint(x);
    t = __builtin_amdgcn_permlane16(u, u, 0x67452301u, 0xEFCDAB89u, false, false); // xor 1
    x = fmaxf(x, __uint_as_float(t)); u = __float_as_uint(x);
    t = __builtin_amdgcn_permlane16(u, u, 0x54761032u, 0xDCFE98BAu, false, false); // xor 2
    x = fmaxf(x, __uint_as_float(t)); u = __float_as_uint(x);
    t = __builtin_amdgcn_permlane16(u, u, 0x32107654u, 0xBA98FEDCu, false, false); // xor 4
    x = fmaxf(x, __uint_as_float(t)); u = __float_as_uint(x);
    t = __builtin_amdgcn_permlane16(u, u, 0xFEDCBA98u, 0x76543210u, false, false); // xor 8
    x = fmaxf(x, __uint_as_float(t));
    return x;
#else
    x = fmaxf(x, __shfl_xor(x, 1, 32));
    x = fmaxf(x, __shfl_xor(x, 2, 32));
    x = fmaxf(x, __shfl_xor(x, 4, 32));
    x = fmaxf(x, __shfl_xor(x, 8, 32));
    return x;
#endif
}

__launch_bounds__(256, 1)
__global__ void fa_fwd_bf16wmma(const float* __restrict__ Q,
                                const float* __restrict__ K,
                                const float* __restrict__ V,
                                float* __restrict__ O)
{
    // Double-buffered staging: one barrier per 64-key block.
    __shared__ __bf16 lK[2][KBLK * D_];          // [buf][key][d]   32 KB
    __shared__ __bf16 lVt[2][D_ * KBLK];         // [buf][d][key]   32 KB
    __shared__ __bf16 lP[WAVES][QROWS * KBLK];   // per-wave P      16 KB

    const int tid  = threadIdx.x;
    const int w    = tid >> 5;
    const int lane = tid & 31;
    const int m    = lane & 15;
    const int hi   = lane >> 4;

    const int bh = blockIdx.y;
    const float* Qg = Q + (size_t)bh * S_ * D_;
    const float* Kg = K + (size_t)bh * S_ * D_;
    const float* Vg = V + (size_t)bh * S_ * D_;
    float*       Og = O + (size_t)bh * S_ * D_;

    const int qbase = blockIdx.x * QBLK + w * QROWS;

    // 1/sqrt(D) * log2(e): scores come out of WMMA already in base-2 domain.
    const float qscale = 0.08838834764831845f * 1.4426950408889634f;

    // ---- Q tile (16 x 128) -> bf16 A-fragments, scale folded in.
    v16bf qa[4];
    {
        const float* qrow = Qg + (size_t)(qbase + m) * D_;
        #pragma unroll
        for (int c = 0; c < 4; ++c) {
            const int b0 = c * 32 + hi * 16;
            const int b1 = c * 32 + hi * 16 + 8;
            #pragma unroll
            for (int e = 0; e < 8; ++e) {
                qa[c][e]     = (__bf16)(qrow[b0 + e] * qscale);
                qa[c][e + 8] = (__bf16)(qrow[b1 + e] * qscale);
            }
        }
    }

    // Constant ones B-fragment: row-sums of P via WMMA.
    v16bf ones;
    #pragma unroll
    for (int e = 0; e < 16; ++e) ones[e] = (__bf16)1.0f;

    // Accumulators (C layout: elem r -> row hi*8+r, col m).
    v8f acc[8] = {};
    v8f l_acc  = {};
    float m_run[8];
    #pragma unroll
    for (int r = 0; r < 8; ++r) m_run[r] = -__builtin_inff();

    // Cooperative staging: thread -> (key row, 32-wide d slice). 64x128 per block.
    const int srow = tid >> 2;        // 0..63
    const int scol = (tid & 3) * 32;  // 0,32,64,96

    v4f kreg[8], vreg[8];

    // ---- Prologue: stage block 0 into buffer 0.
    {
        const float* ks = Kg + (size_t)srow * D_ + scol;
        const float* vs = Vg + (size_t)srow * D_ + scol;
        #pragma unroll
        for (int i = 0; i < 8; ++i) {
            kreg[i] = *(const v4f*)(ks + 4 * i);
            vreg[i] = *(const v4f*)(vs + 4 * i);
        }
        #pragma unroll
        for (int i = 0; i < 8; ++i) {
            #pragma unroll
            for (int j = 0; j < 4; ++j) {
                lK[0][srow * D_ + scol + 4 * i + j]      = (__bf16)kreg[i][j];
                lVt[0][(scol + 4 * i + j) * KBLK + srow] = (__bf16)vreg[i][j];
            }
        }
    }
    __syncthreads();

    const int NKB = S_ / KBLK;  // 32 iterations
    for (int kb = 0; kb < NKB; ++kb) {
        const int cur = kb & 1;
        const __bf16* lKc  = lK[cur];
        const __bf16* lVtc = lVt[cur];

        // ---- Issue block kb+1 global loads now; they fly during compute.
        const bool have_next = (kb + 1 < NKB);
        if (have_next) {
            const float* ks = Kg + (size_t)((kb + 1) * KBLK + srow) * D_ + scol;
            const float* vs = Vg + (size_t)((kb + 1) * KBLK + srow) * D_ + scol;
            #pragma unroll
            for (int i = 0; i < 8; ++i) {
                kreg[i] = *(const v4f*)(ks + 4 * i);
                vreg[i] = *(const v4f*)(vs + 4 * i);
            }
            __builtin_prefetch(ks + KBLK * D_, 0, 0);  // block kb+2 into GL2
            __builtin_prefetch(vs + KBLK * D_, 0, 0);
        }

        // ---- Scores: 4 subtiles of 16 keys x 4 d-chunks = 16 WMMAs.
        v8f s[4] = {};
        #pragma unroll
        for (int c = 0; c < 4; ++c) {
            v16bf kf[4];
            #pragma unroll
            for (int n = 0; n < 4; ++n) {
                const __bf16* kr = &lKc[(n * 16 + m) * D_ + c * 32 + hi * 16];
                #pragma unroll
                for (int e = 0; e < 16; ++e) kf[n][e] = kr[e];
            }
            #pragma unroll
            for (int n = 0; n < 4; ++n)
                s[n] = __builtin_amdgcn_wmma_f32_16x16x32_bf16(false, qa[c], false, kf[n],
                                                               (short)0, s[n], false, false);
        }

        // ---- Online softmax in base-2 domain (rescale amortized over 64 keys).
        float p[4][8];
        #pragma unroll
        for (int r = 0; r < 8; ++r) {
            const float mt    = halfgroup_max(fmaxf(fmaxf(s[0][r], s[1][r]),
                                                    fmaxf(s[2][r], s[3][r])));
            const float mnew  = fmaxf(m_run[r], mt);
            const float alpha = fast_exp2(m_run[r] - mnew);
            m_run[r] = mnew;
            #pragma unroll
            for (int n = 0; n < 4; ++n) p[n][r] = fast_exp2(s[n][r] - mnew);
            l_acc[r] *= alpha;
            #pragma unroll
            for (int t = 0; t < 8; ++t) acc[t][r] *= alpha;
        }

        // ---- Re-layout P (C -> A) through per-wave LDS staging.
        __bf16* pw = &lP[w][0];
        #pragma unroll
        for (int r = 0; r < 8; ++r) {
            #pragma unroll
            for (int n = 0; n < 4; ++n)
                pw[(hi * 8 + r) * KBLK + n * 16 + m] = (__bf16)p[n][r];
        }
        asm volatile("s_wait_dscnt 0x0" ::: "memory"); // wave-local store->load

        v16bf pa[2];  // two 32-key A-fragments
        #pragma unroll
        for (int h = 0; h < 2; ++h) {
            const __bf16* pr = pw + m * KBLK + h * 32;
            #pragma unroll
            for (int e = 0; e < 8; ++e) {
                pa[h][e]     = pr[hi * 16 + e];
                pa[h][e + 8] = pr[hi * 16 + 8 + e];
            }
        }

        // ---- PV: 8 d-tiles x 2 key-halves = 16 WMMAs + 2 ones-WMMAs.
        #pragma unroll
        for (int h = 0; h < 2; ++h) {
            #pragma unroll
            for (int tg = 0; tg < 2; ++tg) {   // groups of 4 tiles bound VGPR pressure
                v16bf vf[4];
                #pragma unroll
                for (int t = 0; t < 4; ++t) {
                    const __bf16* vr = &lVtc[((tg * 4 + t) * 16 + m) * KBLK + h * 32 + hi * 16];
                    #pragma unroll
                    for (int e = 0; e < 16; ++e) vf[t][e] = vr[e];
                }
                #pragma unroll
                for (int t = 0; t < 4; ++t)
                    acc[tg * 4 + t] = __builtin_amdgcn_wmma_f32_16x16x32_bf16(
                        false, pa[h], false, vf[t], (short)0, acc[tg * 4 + t], false, false);
            }
            l_acc = __builtin_amdgcn_wmma_f32_16x16x32_bf16(false, pa[h], false, ones,
                                                            (short)0, l_acc, false, false);
        }

        // ---- Convert & store block kb+1 into the idle buffer (overlaps compute).
        if (have_next) {
            __bf16* lKn  = lK[1 - cur];
            __bf16* lVtn = lVt[1 - cur];
            #pragma unroll
            for (int i = 0; i < 8; ++i) {
                #pragma unroll
                for (int j = 0; j < 4; ++j) {
                    lKn[srow * D_ + scol + 4 * i + j]      = (__bf16)kreg[i][j];
                    lVtn[(scol + 4 * i + j) * KBLK + srow] = (__bf16)vreg[i][j];
                }
            }
        }
        __syncthreads();  // single barrier per 64-key block
    }

    // ---- Epilogue: normalize and store fp32 output.
    #pragma unroll
    for (int r = 0; r < 8; ++r) {
        const float inv = 1.0f / l_acc[r];
        float* orow = Og + (size_t)(qbase + hi * 8 + r) * D_;
        #pragma unroll
        for (int t = 0; t < 8; ++t) orow[t * 16 + m] = acc[t][r] * inv;
    }
}

extern "C" void kernel_launch(void* const* d_in, const int* in_sizes, int n_in,
                              void* d_out, int out_size, void* d_ws, size_t ws_size,
                              hipStream_t stream) {
    (void)in_sizes; (void)n_in; (void)d_ws; (void)ws_size; (void)out_size;
    const float* Q = (const float*)d_in[0];
    const float* K = (const float*)d_in[1];
    const float* V = (const float*)d_in[2];
    float*       O = (float*)d_out;

    dim3 grid(S_ / QBLK, B_ * H_);
    fa_fwd_bf16wmma<<<grid, dim3(256), 0, stream>>>(Q, K, V, O);
}
